// PointNetFeaturePropagation_81587198754954
// MI455X (gfx1250) — matrix operand order repaired and checked by reading, not compiled, not run on hardware
//
#include <hip/hip_runtime.h>
#include <hip/hip_bf16.h>

// ---------- types for WMMA ----------
typedef __attribute__((ext_vector_type(16))) __bf16 v16bf;
typedef __attribute__((ext_vector_type(8)))  float  v8f;

static __device__ __forceinline__ unsigned short f2bf(float f) {
    unsigned int u = __float_as_uint(f);
    unsigned int r = u + 0x7FFFu + ((u >> 16) & 1u);   // round-to-nearest-even
    return (unsigned short)(r >> 16);
}

// ================= kernel 1: zero BN stats =================
__global__ void zero_stats(float* a, float* b, int na, int nb) {
    int t = blockIdx.x * blockDim.x + threadIdx.x;
    if (t < na) a[t] = 0.f;
    if (t < nb) b[t] = 0.f;
}

// ===== kernel 2: pack fp32 weights -> bf16 WMMA-fragment-ordered tiles =====
// Layout: tile = ktile*(C/16) + ntile ; dword index = (tile*32 + lane)*8 + v
// dword holds {W[k,c], W[k+1,c]} with c = ntile*16 + lane%16,
//                                     k = ktile*32 + (v/4)*16 + (lane/16)*8 + (v%4)*2
__global__ void pack_weights_tiled(const float* __restrict__ W, unsigned int* __restrict__ Wt,
                                   int K, int C) {
    int idx = blockIdx.x * blockDim.x + threadIdx.x;
    int total = (K / 2) * C;
    if (idx >= total) return;
    int v     = idx & 7;
    int lane  = (idx >> 3) & 31;
    int tile  = idx >> 8;
    int ntc   = C / 16;
    int ktile = tile / ntc;
    int nt    = tile - ktile * ntc;
    int c = nt * 16 + (lane & 15);
    int k = ktile * 32 + (v >> 2) * 16 + (lane >> 4) * 8 + (v & 3) * 2;
    unsigned int lo = f2bf(W[(size_t)k * C + c]);
    unsigned int hi = f2bf(W[(size_t)(k + 1) * C + c]);
    Wt[idx] = lo | (hi << 16);
}

// ================= kernel 3: 3-NN search (xyz2 staged in LDS, SoA) =================
__global__ void knn3(const float* __restrict__ xyz1, const float* __restrict__ xyz2,
                     int* __restrict__ idx_out, float* __restrict__ w_out,
                     int N, int S) {
    __shared__ float sx[4096], sy[4096], sz[4096];
    int b = blockIdx.y;
    int n = blockIdx.x * blockDim.x + threadIdx.x;
    const float* x2 = xyz2 + (size_t)b * 3 * S;
    for (int i = threadIdx.x; i < S; i += blockDim.x) {
        sx[i] = x2[i];
        sy[i] = x2[S + i];
        sz[i] = x2[2 * S + i];
    }
    __syncthreads();

    const float* x1 = xyz1 + (size_t)b * 3 * N;
    float qx = x1[n], qy = x1[N + n], qz = x1[2 * N + n];

    float d0 = 3.4e38f, d1 = 3.4e38f, d2 = 3.4e38f;
    int   i0 = 0, i1 = 0, i2 = 0;
    for (int s = 0; s < S; ++s) {
        float dx = qx - sx[s], dy = qy - sy[s], dz = qz - sz[s];
        float d = dx * dx + dy * dy + dz * dz;
        if (d < d0)      { d2 = d1; i2 = i1; d1 = d0; i1 = i0; d0 = d; i0 = s; }
        else if (d < d1) { d2 = d1; i2 = i1; d1 = d;  i1 = s; }
        else if (d < d2) { d2 = d;  i2 = s; }
    }
    float r0 = 1.f / fmaxf(d0, 1e-6f);
    float r1 = 1.f / fmaxf(d1, 1e-6f);
    float r2 = 1.f / fmaxf(d2, 1e-6f);
    float inv = 1.f / fmaxf(r0 + r1 + r2, 1e-6f);

    size_t r = (size_t)b * N + n;
    idx_out[r * 3 + 0] = i0;  w_out[r * 3 + 0] = r0 * inv;
    idx_out[r * 3 + 1] = i1;  w_out[r * 3 + 1] = r1 * inv;
    idx_out[r * 3 + 2] = i2;  w_out[r * 3 + 2] = r2 * inv;
}

// ===== kernel 4: transpose points2 [B,D2,S] -> p2t [B,S,D2] via 32x32 LDS tile =====
__global__ void transpose_p2(const float* __restrict__ p2, float* __restrict__ p2t,
                             int S, int D2) {
    __shared__ float tile[32][33];
    int b  = blockIdx.z;
    int s0 = blockIdx.x * 32;
    int c0 = blockIdx.y * 32;
    int t  = threadIdx.x;                 // 256 threads
#pragma unroll
    for (int it = 0; it < 4; ++it) {
        int idx = it * 256 + t;
        int c = idx >> 5, s = idx & 31;   // consecutive t -> consecutive s (coalesced)
        tile[c][s] = p2[((size_t)b * D2 + c0 + c) * S + s0 + s];
    }
    __syncthreads();
#pragma unroll
    for (int it = 0; it < 4; ++it) {
        int idx = it * 256 + t;
        int s = idx >> 5, c = idx & 31;   // consecutive t -> consecutive c (coalesced)
        p2t[((size_t)b * S + s0 + s) * D2 + c0 + c] = tile[c][s];
    }
}

// ===== kernel 5: transpose points1 [B,D1,N] straight into X rows (bf16 pairs) =====
__global__ void transpose_p1_to_x(const float* __restrict__ p1, unsigned int* __restrict__ Xd,
                                  int N, int D1, int Kd /* K/2 dwords per row */) {
    __shared__ float tile[32][33];
    int b  = blockIdx.z;
    int n0 = blockIdx.x * 32;
    int c0 = blockIdx.y * 32;
    int t  = threadIdx.x;                 // 256 threads
#pragma unroll
    for (int it = 0; it < 4; ++it) {
        int idx = it * 256 + t;
        int c = idx >> 5, n = idx & 31;   // coalesced along n
        tile[c][n] = p1[((size_t)b * D1 + c0 + c) * N + n0 + n];
    }
    __syncthreads();
#pragma unroll
    for (int it = 0; it < 2; ++it) {
        int idx = it * 256 + t;
        int row = idx >> 4, dw = idx & 15;   // 16 dwords = 32 channels per row
        int c = dw * 2;
        unsigned int lo = f2bf(tile[c][row]);
        unsigned int hi = f2bf(tile[c + 1][row]);
        Xd[((size_t)b * N + n0 + row) * Kd + (c0 >> 1) + dw] = lo | (hi << 16);
    }
}

// ===== kernel 6: 3-NN weighted gather from p2t (coalesced rows) -> X[:, D1:] ======
__global__ void interp_x(const float* __restrict__ p2t, const int* __restrict__ idx3,
                         const float* __restrict__ wgt3, unsigned int* __restrict__ Xd,
                         int N, int S, int D1, int D2, int Kd) {
    int r = blockIdx.x;                   // 0..B*N-1
    int b = r / N;
    int t = threadIdx.x;                  // 0..127 -> channels 2t, 2t+1
    int   j0 = idx3[(size_t)r * 3 + 0], j1 = idx3[(size_t)r * 3 + 1], j2 = idx3[(size_t)r * 3 + 2];
    float w0 = wgt3[(size_t)r * 3 + 0], w1 = wgt3[(size_t)r * 3 + 1], w2 = wgt3[(size_t)r * 3 + 2];
    const float* q0 = p2t + ((size_t)b * S + j0) * D2;
    const float* q1 = p2t + ((size_t)b * S + j1) * D2;
    const float* q2 = p2t + ((size_t)b * S + j2) * D2;
    int c = 2 * t;
    float v0 = w0 * q0[c]     + w1 * q1[c]     + w2 * q2[c];
    float v1 = w0 * q0[c + 1] + w1 * q1[c + 1] + w2 * q2[c + 1];
    v0 = (v0 == v0 && fabsf(v0) <= 3.0e38f) ? v0 : 0.f;   // nan_to_num
    v1 = (v1 == v1 && fabsf(v1) <= 3.0e38f) ? v1 : 0.f;
    Xd[(size_t)r * Kd + (D1 >> 1) + t] = (unsigned int)f2bf(v0) | ((unsigned int)f2bf(v1) << 16);
}

// ===== kernel 7: bf16 WMMA GEMM, 32x64 output per wave, 4 waves per block ==========
// Per k-chunk: 4 A b128-loads + 8 B b128-loads feed 8 v_wmma (1.5 loads/WMMA).
template <int KTOT, int COUT>
__global__ void __launch_bounds__(128)
gemm_bf16(const unsigned short* __restrict__ A,   // [M, KTOT] bf16 row-major
          const unsigned int*   __restrict__ Wt,  // fragment-ordered tiles
          const float* __restrict__ bias,
          float* __restrict__ Y) {
    constexpr int NT = COUT / 16;
    const int lane  = threadIdx.x & 31;
    const int wave  = threadIdx.x >> 5;
    const int row0  = (blockIdx.x * 4 + wave) * 32;   // 32 rows per wave
    const int ct0   = blockIdx.y * 4;                 // first of 4 column tiles
    const int l16   = lane & 15;
    const int khalf = lane >> 4;
    const int m0    = row0 + l16;
    const int m1    = m0 + 16;

    v8f acc[2][4] = {};

    for (int kc = 0; kc < KTOT; kc += 32) {
        // Two A fragments (rows m0.., m1..): 2x contiguous b128 loads each
        union { unsigned int u[8]; v16bf v; } af0, af1;
        {
            const unsigned int* pa =
                (const unsigned int*)(A + (size_t)m0 * KTOT + kc + khalf * 8);
#pragma unroll
            for (int i = 0; i < 4; ++i) af0.u[i] = pa[i];
            const unsigned int* pa2 =
                (const unsigned int*)(A + (size_t)m0 * KTOT + kc + 16 + khalf * 8);
#pragma unroll
            for (int i = 0; i < 4; ++i) af0.u[4 + i] = pa2[i];
        }
        {
            const unsigned int* pa =
                (const unsigned int*)(A + (size_t)m1 * KTOT + kc + khalf * 8);
#pragma unroll
            for (int i = 0; i < 4; ++i) af1.u[i] = pa[i];
            const unsigned int* pa2 =
                (const unsigned int*)(A + (size_t)m1 * KTOT + kc + 16 + khalf * 8);
#pragma unroll
            for (int i = 0; i < 4; ++i) af1.u[4 + i] = pa2[i];
        }

        // 4 B fragments, each shared by both row fragments
        const unsigned int* pb =
            Wt + ((size_t)((kc >> 5) * NT + ct0) * 32 + lane) * 8;
#pragma unroll
        for (int j = 0; j < 4; ++j) {
            union { unsigned int u[8]; v16bf v; } bf;
#pragma unroll
            for (int i = 0; i < 8; ++i) bf.u[i] = pb[j * 256 + i];
            acc[0][j] = __builtin_amdgcn_wmma_f32_16x16x32_bf16(
                false, af0.v, false, bf.v, (short)0, acc[0][j], false, false);
            acc[1][j] = __builtin_amdgcn_wmma_f32_16x16x32_bf16(
                false, af1.v, false, bf.v, (short)0, acc[1][j], false, false);
        }
    }

#pragma unroll
    for (int h = 0; h < 2; ++h) {
        const int mrow = row0 + h * 16 + khalf * 8;
#pragma unroll
        for (int j = 0; j < 4; ++j) {
            const int col = (ct0 + j) * 16 + l16;
            const float bv = bias[col];
#pragma unroll
            for (int i = 0; i < 8; ++i)
                Y[(size_t)(mrow + i) * COUT + col] = acc[h][j][i] + bv;
        }
    }
}

// ================= kernel 8: per-channel sum / sumsq partials =================
template <int C>
__global__ void bn_stats(const float* __restrict__ Y, float* __restrict__ stats,
                         int Mtot, int rowsPerBlock) {
    int c  = threadIdx.x;                 // blockDim.x == C
    int r0 = blockIdx.x * rowsPerBlock;
    int r1 = r0 + rowsPerBlock; if (r1 > Mtot) r1 = Mtot;
    float s = 0.f, s2 = 0.f;
    for (int r = r0; r < r1; ++r) {
        float y = Y[(size_t)r * C + c];
        s += y; s2 += y * y;
    }
    atomicAdd(&stats[c],     s);
    atomicAdd(&stats[C + c], s2);
}

// ================= kernel 9: BN + ReLU -> bf16 (feeds GEMM2) =================
__global__ void bn_relu_bf16(const float* __restrict__ Y, const float* __restrict__ stats,
                             const float* __restrict__ g, const float* __restrict__ beta,
                             unsigned short* __restrict__ H, int Mtot, int C) {
    int r = blockIdx.x, c = threadIdx.x;
    float invM = 1.f / (float)Mtot;
    float mean = stats[c] * invM;
    float var  = stats[C + c] * invM - mean * mean;
    float sc   = g[c] * rsqrtf(var + 1e-5f);
    float v    = sc * (Y[(size_t)r * C + c] - mean) + beta[c];
    v = fmaxf(v, 0.f);
    H[(size_t)r * C + c] = f2bf(v);
}

// ============ kernel 10: BN + ReLU + transpose to [B, C2, N] fp32 output ============
__global__ void bn_relu_out(const float* __restrict__ Y, const float* __restrict__ stats,
                            const float* __restrict__ g, const float* __restrict__ beta,
                            float* __restrict__ out, int Mtot, int C, int N) {
    int n = blockIdx.x * blockDim.x + threadIdx.x;
    int c = blockIdx.y;
    int b = blockIdx.z;
    int r = b * N + n;
    float invM = 1.f / (float)Mtot;
    float mean = stats[c] * invM;
    float var  = stats[C + c] * invM - mean * mean;
    float sc   = g[c] * rsqrtf(var + 1e-5f);
    float v    = sc * (Y[(size_t)r * C + c] - mean) + beta[c];
    v = fmaxf(v, 0.f);
    v = (v == v && fabsf(v) <= 3.0e38f) ? v : 0.f;     // nan_to_num
    out[((size_t)b * C + c) * N + n] = v;
}

// ======================================================================================
extern "C" void kernel_launch(void* const* d_in, const int* in_sizes, int n_in,
                              void* d_out, int out_size, void* d_ws, size_t ws_size,
                              hipStream_t stream) {
    const float* xyz1    = (const float*)d_in[0];
    const float* xyz2    = (const float*)d_in[1];
    const float* points1 = (const float*)d_in[2];
    const float* points2 = (const float*)d_in[3];
    const float* w1      = (const float*)d_in[4];
    const float* b1      = (const float*)d_in[5];
    const float* g1      = (const float*)d_in[6];
    const float* be1     = (const float*)d_in[7];
    const float* w2      = (const float*)d_in[8];
    const float* b2      = (const float*)d_in[9];
    const float* g2      = (const float*)d_in[10];
    const float* be2     = (const float*)d_in[11];
    float* out = (float*)d_out;

    constexpr int B = 2, N = 16384, S = 4096, D1 = 128, D2 = 256, C1 = 256, C2 = 128;
    constexpr int K1 = D1 + D2;           // 384
    constexpr int Kd = K1 / 2;            // 192 dwords per X row
    constexpr int M  = B * N;             // 32768

    char* ws = (char*)d_ws;
    size_t off = 0;
    auto alloc = [&](size_t bytes) {
        size_t o = off;
        off += (bytes + 255) & ~(size_t)255;
        return o;
    };
    unsigned int*   w1t  = (unsigned int*)  (ws + alloc((size_t)(K1 / 2) * C1 * 4));
    unsigned int*   w2t  = (unsigned int*)  (ws + alloc((size_t)(C1 / 2) * C2 * 4));
    int*            idx3 = (int*)           (ws + alloc((size_t)M * 3 * 4));
    float*          wgt3 = (float*)         (ws + alloc((size_t)M * 3 * 4));
    float*          p2t  = (float*)         (ws + alloc((size_t)B * S * D2 * 4));
    unsigned int*   Xd   = (unsigned int*)  (ws + alloc((size_t)M * Kd * 4));
    float*          Y1   = (float*)         (ws + alloc((size_t)M * C1 * 4));
    float*          st1  = (float*)         (ws + alloc((size_t)2 * C1 * 4));
    unsigned short* H1   = (unsigned short*)(ws + alloc((size_t)M * C1 * 2));
    float*          Y2   = (float*)         (ws + alloc((size_t)M * C2 * 4));
    float*          st2  = (float*)         (ws + alloc((size_t)2 * C2 * 4));

    // stats must be zero before atomics accumulate (graph replays reuse ws)
    zero_stats<<<1, 2 * C1, 0, stream>>>(st1, st2, 2 * C1, 2 * C2);

    pack_weights_tiled<<<((K1 / 2) * C1 + 255) / 256, 256, 0, stream>>>(w1, w1t, K1, C1);
    pack_weights_tiled<<<((C1 / 2) * C2 + 255) / 256, 256, 0, stream>>>(w2, w2t, C1, C2);

    dim3 gknn(N / 256, B);
    knn3<<<gknn, 256, 0, stream>>>(xyz1, xyz2, idx3, wgt3, N, S);

    dim3 gtp2(S / 32, D2 / 32, B);
    transpose_p2<<<gtp2, 256, 0, stream>>>(points2, p2t, S, D2);

    dim3 gtp1(N / 32, D1 / 32, B);
    transpose_p1_to_x<<<gtp1, 256, 0, stream>>>(points1, Xd, N, D1, Kd);

    interp_x<<<M, 128, 0, stream>>>(p2t, idx3, wgt3, Xd, N, S, D1, D2, Kd);

    dim3 gg1(M / 128, C1 / 64);
    gemm_bf16<K1, C1><<<gg1, 128, 0, stream>>>((const unsigned short*)Xd, w1t, b1, Y1);

    bn_stats<C1><<<256, C1, 0, stream>>>(Y1, st1, M, M / 256);
    bn_relu_bf16<<<M, C1, 0, stream>>>(Y1, st1, g1, be1, H1, M, C1);

    dim3 gg2(M / 128, C2 / 64);
    gemm_bf16<C1, C2><<<gg2, 128, 0, stream>>>(H1, w2t, b2, Y2);

    bn_stats<C2><<<256, C2, 0, stream>>>(Y2, st2, M, M / 256);

    dim3 go(N / 256, C2, B);
    bn_relu_out<<<go, 256, 0, stream>>>(Y2, st2, g2, be2, out, M, C2, N);
}
